// involution_18743237280772
// MI455X (gfx1250) — matrix operand order repaired and checked by reading, not compiled, not run on hardware
//
#include <hip/hip_runtime.h>

#define B_   4
#define C_   256
#define H_   128
#define W_   128
#define CR_  64
#define KK_  9
#define EPS_ 1e-5f

typedef __attribute__((ext_vector_type(16))) __bf16 v16bf;
typedef __attribute__((ext_vector_type(8)))  float  v8f;

// LDS strides (in elements). Row byte-starts are multiples of 16 for b128 loads.
#define XSS 264   // bf16 stride for 256-wide rows (528B = 33*16)
#define H1S 72    // bf16 stride for 64-wide rows  (144B = 9*16)
#define W2S 72
#define WKS 149   // f32 stride; 149 mod 64 coprime with 64 -> conflict-free

__device__ __forceinline__ unsigned short f2bf(float f) {
  unsigned u = __float_as_uint(f);
  u += 0x7FFFu + ((u >> 16) & 1u);          // round-to-nearest-even
  return (unsigned short)(u >> 16);
}

union FragBF { v16bf v; uint4 q[2]; };

__global__ __launch_bounds__(256)
void involution_fused(const float* __restrict__ x,
                      const float* __restrict__ w1,
                      const float* __restrict__ gma,
                      const float* __restrict__ bta,
                      const float* __restrict__ mu,
                      const float* __restrict__ var,
                      const float* __restrict__ w2,
                      const float* __restrict__ b2,
                      float* __restrict__ out) {
  __shared__ __align__(16) unsigned short xsA[32 * XSS];   // x strip   [px][c]   bf16
  __shared__ __align__(16) unsigned short w1s[64 * XSS];   // BN-folded [co][ci]  bf16
  __shared__ __align__(16) unsigned short h1s[32 * H1S];   // h1        [px][co]  bf16
  __shared__ __align__(16) unsigned short w2s[144 * W2S];  // w2        [o][ci]   bf16
  __shared__ __align__(16) float          wks[32 * WKS];   // wk        [px][o]   f32

  const int tid  = threadIdx.x;
  const int wave = tid >> 5;
  const int lane = tid & 31;
  const int x0 = blockIdx.x * 32;
  const int y  = blockIdx.y;
  const int b  = blockIdx.z;

  // ---------------- Phase 1: stage inputs into LDS (bf16) ----------------
  {
    // thread == channel: 32 contiguous f32 per channel row (coalesced 128B/lane)
    const float* xr = x + (((size_t)b * C_ + tid) * H_ + y) * W_ + x0;
    #pragma unroll 8
    for (int p = 0; p < 32; ++p)
      xsA[p * XSS + tid] = f2bf(xr[p]);

    // w1 with BN scale folded into rows
    for (int i = tid; i < CR_ * C_; i += 256) {
      int r = i >> 8, k = i & 255;
      float s = gma[r] * rsqrtf(var[r] + EPS_);
      w1s[r * XSS + k] = f2bf(w1[i] * s);
    }
    for (int i = tid; i < 144 * 64; i += 256) {
      int r = i >> 6, k = i & 63;
      w2s[r * W2S + k] = f2bf(w2[i]);
    }
  }
  __syncthreads();

  // ---------------- Phase 2: GEMM1  h1 = relu(x*w1' + b1)  (M=32,N=64,K=256)
  {
    const int mt = wave >> 2, nt = wave & 3;       // one 16x16 tile per wave
    const int hi = (lane >= 16) ? 1 : 0;
    const int aoff = (mt * 16 + (lane & 15)) * XSS + hi * 8;   // A: row=pixel
    const int co   = nt * 16 + (lane & 15);
    const int boff = co * XSS + hi * 8;                         // B: col=channel
    v8f acc = {};
    #pragma unroll
    for (int ks = 0; ks < 8; ++ks) {               // K = 8 * 32
      FragBF a, bf;
      a.q[0]  = *(const uint4*)&xsA[aoff + ks * 32];
      a.q[1]  = *(const uint4*)&xsA[aoff + ks * 32 + 16];
      bf.q[0] = *(const uint4*)&w1s[boff + ks * 32];
      bf.q[1] = *(const uint4*)&w1s[boff + ks * 32 + 16];
      acc = __builtin_amdgcn_wmma_f32_16x16x32_bf16(false, a.v, false, bf.v,
                                                    (short)0, acc, false, false);
    }
    const float bias = bta[co] - mu[co] * (gma[co] * rsqrtf(var[co] + EPS_));
    #pragma unroll
    for (int r = 0; r < 8; ++r) {
      int p = mt * 16 + r + hi * 8;
      float h = fmaxf(acc[r] + bias, 0.0f);
      h1s[p * H1S + co] = f2bf(h);
    }
  }
  __syncthreads();

  // ---------------- Phase 3: GEMM2  wk = h1*w2' + b2  (M=32,N=144,K=64)
  for (int t = wave; t < 18; t += 8) {             // uniform scalar loop per wave
    const int mt = t & 1, nt = t >> 1;
    const int hi = (lane >= 16) ? 1 : 0;
    const int aoff = (mt * 16 + (lane & 15)) * H1S + hi * 8;
    const int n    = nt * 16 + (lane & 15);        // n in [0,144)
    const int boff = n * W2S + hi * 8;
    v8f acc = {};
    #pragma unroll
    for (int ks = 0; ks < 2; ++ks) {
      FragBF a, bf;
      a.q[0]  = *(const uint4*)&h1s[aoff + ks * 32];
      a.q[1]  = *(const uint4*)&h1s[aoff + ks * 32 + 16];
      bf.q[0] = *(const uint4*)&w2s[boff + ks * 32];
      bf.q[1] = *(const uint4*)&w2s[boff + ks * 32 + 16];
      acc = __builtin_amdgcn_wmma_f32_16x16x32_bf16(false, a.v, false, bf.v,
                                                    (short)0, acc, false, false);
    }
    const float bias = b2[n];
    #pragma unroll
    for (int r = 0; r < 8; ++r) {
      int p = mt * 16 + r + hi * 8;
      wks[p * WKS + n] = acc[r] + bias;
    }
  }
  __syncthreads();

  // ---------------- Phase 4: involution (lane = x position; coalesced rows)
  {
    const int gx = x0 + lane;
    for (int cc = 0; cc < 32; ++cc) {
      const int c = wave * 32 + cc;                // 8 waves * 32 ch = 256
      const int g = c >> 4;
      const float* xc = x + ((size_t)b * C_ + c) * (size_t)(H_ * W_);
      float acc = 0.0f;
      #pragma unroll
      for (int dy = -1; dy <= 1; ++dy) {
        const int yy = y + dy;
        if (yy < 0 || yy >= H_) continue;          // uniform (y is per-block)
        const float* row = xc + (size_t)yy * W_;
        const int t0 = (dy + 1) * 3;
        const float wm = wks[lane * WKS + g * KK_ + t0 + 0];
        const float w0 = wks[lane * WKS + g * KK_ + t0 + 1];
        const float wp = wks[lane * WKS + g * KK_ + t0 + 2];
        float v0 = row[gx];
        int ixm = (gx == 0) ? 0 : gx - 1;
        float vm = row[ixm];  vm = (gx == 0) ? 0.0f : vm;
        int ixp = (gx == W_ - 1) ? W_ - 1 : gx + 1;
        float vp = row[ixp];  vp = (gx == W_ - 1) ? 0.0f : vp;
        acc = fmaf(wm, vm, acc);
        acc = fmaf(w0, v0, acc);
        acc = fmaf(wp, vp, acc);
      }
      out[(((size_t)b * C_ + c) * H_ + y) * W_ + gx] = acc;
    }
  }
}

extern "C" void kernel_launch(void* const* d_in, const int* in_sizes, int n_in,
                              void* d_out, int out_size, void* d_ws, size_t ws_size,
                              hipStream_t stream) {
  const float* x   = (const float*)d_in[0];
  const float* w1  = (const float*)d_in[1];
  const float* gm  = (const float*)d_in[2];
  const float* bt  = (const float*)d_in[3];
  const float* mu  = (const float*)d_in[4];
  const float* vr  = (const float*)d_in[5];
  const float* w2  = (const float*)d_in[6];
  const float* b2  = (const float*)d_in[7];
  dim3 grid(W_ / 32, H_, B_);
  involution_fused<<<grid, 256, 0, stream>>>(x, w1, gm, bt, mu, vr, w2, b2,
                                             (float*)d_out);
}
